// Conv2DAttention_26766236189067
// MI455X (gfx1250) — matrix operand order repaired and checked by reading, not compiled, not run on hardware
//
#include <hip/hip_runtime.h>
#include <hip/hip_bf16.h>
#include <stdint.h>

// Problem constants (fixed by reference)
#define NHEAD 8
#define DH    8
#define CCH   64
#define BATCH 4
#define HDIM  256
#define WDIM  256
#define HWPIX (HDIM * WDIM)            // 65536
#define NPIX  (BATCH * HWPIX)          // 262144

typedef __attribute__((ext_vector_type(16))) __bf16 v16bf;
typedef __attribute__((ext_vector_type(8)))  __bf16 v8bf;
typedef __attribute__((ext_vector_type(4)))  __bf16 v4bf;
typedef __attribute__((ext_vector_type(8)))  float  v8f;
typedef __attribute__((ext_vector_type(4)))  float  v4f;

__device__ __forceinline__ v16bf join16(v8bf lo, v8bf hi) {
  return __builtin_shufflevector(lo, hi, 0, 1, 2, 3, 4, 5, 6, 7,
                                 8, 9, 10, 11, 12, 13, 14, 15);
}

// CDNA5 async global->LDS copy (16B per lane, tracked on ASYNCcnt).
__device__ __forceinline__ void async_copy_b128(unsigned ldsOff, const void* gaddr) {
  asm volatile("global_load_async_to_lds_b128 %0, %1, off"
               :: "v"(ldsOff), "v"((unsigned long long)(size_t)gaddr)
               : "memory");
}
__device__ __forceinline__ void wait_async() {
  asm volatile("s_wait_asynccnt 0x0" ::: "memory");
}

// ---------------------------------------------------------------------------
// Kernel 1: Q/K/V projections, computed transposed: (X W)^T = W^T X^T.
//   M = out-channel, N = pixel, K = in-channel.
//   - image tile async-copied to LDS, then transpose-converted to bf16 [p][c]
//   - B fragments (image) built once per wave, shared by all 3 weights
//   - D lane holds one pixel with 8 consecutive channels -> packed b128 stores
// Per wave: one 16-pixel tile, 24 x v_wmma_f32_16x16x32_bf16.
// ---------------------------------------------------------------------------
__launch_bounds__(256)
__global__ void qkv_proj_kernel(const float* __restrict__ img,
                                const float* __restrict__ Wq,
                                const float* __restrict__ Wk,
                                const float* __restrict__ Wv,
                                __bf16* __restrict__ Qb,
                                __bf16* __restrict__ Kb,
                                __bf16* __restrict__ Vb) {
  __shared__ __align__(16) float  imgT[CCH][128];      // 32 KB: [channel][pixel] fp32
  __shared__ __align__(16) __bf16 xT[128][CCH];        // 16 KB: [pixel][channel] bf16
  __shared__ __align__(16) __bf16 wT[3][CCH][CCH];     // 24 KB: [w][outch][inch]

  const int tid  = threadIdx.x;
  const int wave = tid >> 5;
  const int lane = tid & 31;
  const int lm   = lane & 15;
  const bool hiL = lane >= 16;

  const int gblk  = blockIdx.x * 128;                  // block pixel base
  const int batch = gblk / HWPIX;
  const float* ib = img + (size_t)batch * CCH * HWPIX + (gblk % HWPIX);

  // Async-stage the 128px x 64ch fp32 tile: wave w copies channels 8w..8w+7,
  // one coalesced 512B row per channel (32 lanes x b128).
#pragma unroll
  for (int ci = 0; ci < 8; ++ci) {
    const int c = wave * 8 + ci;
    async_copy_b128((unsigned)(size_t)&imgT[c][lane * 4],
                    ib + (size_t)c * HWPIX + lane * 4);
  }

  // Stage weights transposed to bf16: wT[w][outch][inch] = W[inch*64+outch].
  for (int i = tid; i < CCH * CCH; i += 256) {
    const int k = i >> 6, n = i & 63;
    wT[0][n][k] = (__bf16)Wq[i];
    wT[1][n][k] = (__bf16)Wk[i];
    wT[2][n][k] = (__bf16)Wv[i];
  }

  wait_async();
  __syncthreads();

  // Transpose-convert image tile: fp32 [c][p] -> bf16 [p][c] (b64 stores).
  for (int idx = tid; idx < 128 * (CCH / 4); idx += 256) {
    const int p  = idx >> 4;
    const int c4 = (idx & 15) * 4;
    v4bf t;
    t[0] = (__bf16)imgT[c4 + 0][p];
    t[1] = (__bf16)imgT[c4 + 1][p];
    t[2] = (__bf16)imgT[c4 + 2][p];
    t[3] = (__bf16)imgT[c4 + 3][p];
    *(v4bf*)&xT[p][c4] = t;
  }
  __syncthreads();

  // B fragments (X^T, 32in x 16px), built once, reused by all weights/M-tiles.
  const int pix = wave * 16 + lm;                      // lane's pixel (N column)
  v16bf bfr[2];
#pragma unroll
  for (int kt = 0; kt < 2; ++kt) {
    const int kb = kt * 32 + (hiL ? 8 : 0);
    bfr[kt] = join16(*(const v8bf*)&xT[pix][kb],
                     *(const v8bf*)&xT[pix][kb + 16]);
  }

  const int g0 = gblk + wave * 16;                     // wave's 16-pixel tile
#pragma unroll
  for (int w = 0; w < 3; ++w) {
    __bf16* dst = (w == 0) ? Qb : (w == 1) ? Kb : Vb;
#pragma unroll
    for (int mt = 0; mt < 4; ++mt) {
      const int oc = mt * 16 + lm;                     // A-row (out channel)
      v8f acc = {};
#pragma unroll
      for (int kt = 0; kt < 2; ++kt) {
        // A fragment (W^T, 16out x 32in): per-lane k runs of 8, contiguous.
        const int ka = kt * 32 + (hiL ? 8 : 0);
        const v16bf afr = join16(*(const v8bf*)&wT[w][oc][ka],
                                 *(const v8bf*)&wT[w][oc][ka + 16]);
        acc = __builtin_amdgcn_wmma_f32_16x16x32_bf16(
            false, afr, false, bfr[kt], (short)0, acc, false, false);
      }
      // D': lane = pixel g0+lm, VGPR r = channel mt*16+(hi?8:0)+r -> contiguous
      v8bf dv;
#pragma unroll
      for (int r = 0; r < 8; ++r) dv[r] = (__bf16)acc[r];
      *(v8bf*)(dst + (size_t)(g0 + lm) * CCH + mt * 16 + (hiL ? 8 : 0)) = dv;
    }
  }
}

// ---------------------------------------------------------------------------
// Kernel 2: circular 3x3 attention (VALU) + output projection (WMMA).
// Block = 256 threads = 128 pixels; attention result staged in LDS as bf16.
// O-proj keeps lane=channel / consecutive-pixels orientation because the
// final output is channel-major -> float4 (b128) stores.
// ---------------------------------------------------------------------------
__launch_bounds__(256)
__global__ void attn_oproj_kernel(const __bf16* __restrict__ Qb,
                                  const __bf16* __restrict__ Kb,
                                  const __bf16* __restrict__ Vb,
                                  const float* __restrict__ Wo,
                                  float* __restrict__ out) {
  __shared__ __align__(16) __bf16 attnLds[128][CCH];   // 16 KB, [pixel][channel]
  __shared__ __align__(16) __bf16 woT[CCH][CCH];       //  8 KB, [n][k] (transposed)
  const int tid = threadIdx.x;

  for (int i = tid; i < CCH * CCH; i += 256)
    woT[i & 63][i >> 6] = (__bf16)Wo[i];

  const int g0 = blockIdx.x * 128;
  const float scale = 0.35355339059327373f;            // 1/sqrt(dh)

#pragma unroll
  for (int t = 0; t < 4; ++t) {
    const int task = tid + t * 256;                    // (pixel, head) task
    const int pl   = task >> 3;
    const int head = task & 7;
    const int g  = g0 + pl;
    const int bb = g >> 16;                            // / HWPIX
    const int p  = g & (HWPIX - 1);
    const int h  = p >> 8;                             // / WDIM
    const int ww = p & (WDIM - 1);

    // query slice (8 bf16, 16B aligned)
    const v8bf qv = *(const v8bf*)(Qb + (size_t)g * CCH + head * DH);
    float qf[DH];
#pragma unroll
    for (int d = 0; d < DH; ++d) qf[d] = (float)qv[d];

    int   ngs[8];
    float sc[8];
    float mx = -1e30f;
    int j = 0;
#pragma unroll
    for (int ay = 0; ay < 3; ++ay)
#pragma unroll
      for (int ax = 0; ax < 3; ++ax) {
        if (ay == 1 && ax == 1) continue;
        const int hh = (h + ay - 1) & (HDIM - 1);      // circular wrap
        const int wn = (ww + ax - 1) & (WDIM - 1);
        const int ng = bb * HWPIX + hh * WDIM + wn;
        ngs[j] = ng;
        const v8bf kv = *(const v8bf*)(Kb + (size_t)ng * CCH + head * DH);
        float s = 0.f;
#pragma unroll
        for (int d = 0; d < DH; ++d) s += qf[d] * (float)kv[d];
        sc[j] = s * scale;
        mx = fmaxf(mx, sc[j]);
        ++j;
      }

    float e[8], sum = 0.f;
#pragma unroll
    for (j = 0; j < 8; ++j) { e[j] = __expf(sc[j] - mx); sum += e[j]; }
    const float inv = 1.f / sum;

    float o[DH] = {};
#pragma unroll
    for (j = 0; j < 8; ++j) {
      const v8bf vv = *(const v8bf*)(Vb + (size_t)ngs[j] * CCH + head * DH);
      const float wj = e[j];
#pragma unroll
      for (int d = 0; d < DH; ++d) o[d] += wj * (float)vv[d];
    }
    v8bf ov;
#pragma unroll
    for (int d = 0; d < DH; ++d) ov[d] = (__bf16)(o[d] * inv);
    *(v8bf*)&attnLds[pl][head * DH] = ov;              // one ds_store_b128
  }
  __syncthreads();

  // ---- output projection: attn[128,64] (LDS, bf16) x Wo[64,64] -> out fp32
  const int wave = tid >> 5;
  const int lane = tid & 31;
  const int lm   = lane & 15;
  const bool hiL = lane >= 16;
  const int m0   = wave * 16;

  // A fragment halves are contiguous in [pixel][channel] layout: k 0..7 / 16..23.
  v16bf a[2];
#pragma unroll
  for (int kt = 0; kt < 2; ++kt) {
    const int kb0 = kt * 32 + (hiL ? 8 : 0);
    a[kt] = join16(*(const v8bf*)&attnLds[m0 + lm][kb0],
                   *(const v8bf*)&attnLds[m0 + lm][kb0 + 16]);
  }

  const int g  = g0 + m0;
  const int bb = g >> 16;
  const int p  = g & (HWPIX - 1);
  float* ob = out + (size_t)bb * CCH * HWPIX + p;

#pragma unroll
  for (int nt = 0; nt < 4; ++nt) {
    const int n = nt * 16 + lm;
    v8f acc = {};
#pragma unroll
    for (int kt = 0; kt < 2; ++kt) {
      const int kb = kt * 32 + (hiL ? 8 : 0);
      acc = __builtin_amdgcn_wmma_f32_16x16x32_bf16(
          false, a[kt], false,
          join16(*(const v8bf*)&woT[n][kb], *(const v8bf*)&woT[n][kb + 16]),
          (short)0, acc, false, false);
    }
    // lane = channel n, VGPR r = consecutive pixels -> two b128 stores.
    const int mB = hiL ? 8 : 0;
    float* dst = ob + (size_t)n * HWPIX + mB;
    v4f s0 = {acc[0], acc[1], acc[2], acc[3]};
    v4f s1 = {acc[4], acc[5], acc[6], acc[7]};
    *(v4f*)dst       = s0;
    *(v4f*)(dst + 4) = s1;
  }
}

// ---------------------------------------------------------------------------
// Launch: inputs {image, Wq, Wk, Wv, Wo}, output [B,C,H,W] fp32.
// Workspace: Q/K/V in bf16, 3 * 262144 * 64 * 2B = 100.7 MB.
// ---------------------------------------------------------------------------
extern "C" void kernel_launch(void* const* d_in, const int* in_sizes, int n_in,
                              void* d_out, int out_size, void* d_ws, size_t ws_size,
                              hipStream_t stream) {
  const float* img = (const float*)d_in[0];
  const float* Wq  = (const float*)d_in[1];
  const float* Wk  = (const float*)d_in[2];
  const float* Wv  = (const float*)d_in[3];
  const float* Wo  = (const float*)d_in[4];
  float* out = (float*)d_out;

  const size_t n64 = (size_t)NPIX * CCH;
  __bf16* Qb = (__bf16*)d_ws;
  __bf16* Kb = Qb + n64;
  __bf16* Vb = Kb + n64;

  const int blocks = NPIX / 128;                       // 2048
  qkv_proj_kernel<<<blocks, 256, 0, stream>>>(img, Wq, Wk, Wv, Qb, Kb, Vb);
  attn_oproj_kernel<<<blocks, 256, 0, stream>>>(Qb, Kb, Vb, Wo, out);
}
